// Mamba_42185168781537
// MI455X (gfx1250) — matrix operand order repaired and checked by reading, not compiled, not run on hardware
//
#include <hip/hip_runtime.h>

// ---------------------------------------------------------------------------
// Mamba-style SSM for MI455X (gfx1250), bf16 WMMA pipeline.
// B=16 L=128 O=128 N=256 S=64 U=32 DEL=64 HID=512 K=4 D_OUT=24640
// ---------------------------------------------------------------------------

typedef unsigned short u16;
typedef __attribute__((ext_vector_type(16))) __bf16 v16bf;
typedef __attribute__((ext_vector_type(8)))  float  v8f;

static __device__ __forceinline__ float bf2f(u16 h) {
  unsigned u = ((unsigned)h) << 16;
  return __builtin_bit_cast(float, u);
}
static __device__ __forceinline__ u16 f2bf(float f) {
  unsigned u = __builtin_bit_cast(unsigned, f);
  unsigned r = u + 0x7FFFu + ((u >> 16) & 1u);   // round-to-nearest-even
  return (u16)(r >> 16);
}
static __device__ __forceinline__ float silu_f(float v) {
  return v / (1.0f + __expf(-v));
}

// ---- WMMA fragment loads (contiguous-k order, k = 16*(lane>>4)+i, wave32) ---------------
static __device__ __forceinline__ v16bf load_a_frag(const u16* Als, int mbase, int lane) {
  // A 16x32 (MxK): lane holds row m=lane&15; elems i <-> K = 16*(lane>>4)+i
  int m = lane & 15, g = (lane >> 4) & 1;
  const u16* p0 = Als + (mbase + m) * 32 + 16 * g;
  const u16* p1 = Als + (mbase + m) * 32 + 16 * g + 8;
  v16bf a;
#pragma unroll
  for (int i = 0; i < 8; ++i) a[i] = __builtin_bit_cast(__bf16, p0[i]);
#pragma unroll
  for (int i = 0; i < 8; ++i) a[8 + i] = __builtin_bit_cast(__bf16, p1[i]);
  return a;
}
static __device__ __forceinline__ v16bf load_b_frag(const u16* p) {
  // B 32x16 (KxN): lane holds column n=lane&15; elems i <-> K = 16*(lane>>4)+i
  v16bf b;
#pragma unroll
  for (int i = 0; i < 16; ++i) b[i] = __builtin_bit_cast(__bf16, p[i]);
  return b;
}

// ---- generic conversion kernels -------------------------------------------
__global__ void f2bf_k(const float* __restrict__ src, u16* __restrict__ dst, int n) {
  int i = blockIdx.x * blockDim.x + threadIdx.x;
  int stride = gridDim.x * blockDim.x;
  for (; i < n; i += stride) dst[i] = f2bf(src[i]);
}
// gather x[:, 0:128, :] -> token-major bf16 (2048 x 64)
__global__ void xtok_k(const float* __restrict__ x, u16* __restrict__ dst) {
  int i = blockIdx.x * blockDim.x + threadIdx.x;   // 131072 total
  if (i >= 2048 * 64) return;
  int m = i >> 6, s = i & 63;
  int b = m >> 7, o = m & 127;
  dst[i] = f2bf(x[(size_t)b * 16384 + o * 64 + s]);
}

// ---- bf16 WMMA GEMM:  C(MxN) = A(MxK) @ W(NxK)^T + bias, fused epilogue ---
// block: 256 thr (8 waves). Macro tile 64(M) x 256(N); wave tile 64x32.
// Double-buffered LDS staging of A; scalar (SGPR) tail guards so EXEC is
// untouched around v_wmma; global_prefetch of next weight k-tile.
template <int EPI>
__global__ __launch_bounds__(256) void gemm_bf16_k(
    const u16* __restrict__ A, const u16* __restrict__ Bw,
    const float* __restrict__ bias,
    float* __restrict__ outF, u16* __restrict__ outB0, u16* __restrict__ outB1,
    int M, int N, int K)
{
  __shared__ u16 Als[2][64 * 32];
  const int tid  = threadIdx.x;
  const int lane = tid & 31;
  const int m0   = blockIdx.y * 64;
  // wave-tile N base, made explicitly scalar for uniform control flow
  const int n0w  = __builtin_amdgcn_readfirstlane(blockIdx.x * 256 + (tid >> 5) * 32);

  v8f acc[4][2];
  v8f zero = {0.f, 0.f, 0.f, 0.f, 0.f, 0.f, 0.f, 0.f};
#pragma unroll
  for (int mt = 0; mt < 4; ++mt)
#pragma unroll
    for (int nt = 0; nt < 2; ++nt) acc[mt][nt] = zero;

  const int nk = K >> 5;

  // stage k-tile kt of the 64xK A panel into LDS buffer `buf`
  auto stage = [&](int kt, int buf) {
#pragma unroll
    for (int j = 0; j < 8; ++j) {
      int idx = tid + j * 256;
      int r = idx >> 5, c = idx & 31;
      Als[buf][idx] = A[(size_t)(m0 + r) * K + kt * 32 + c];
    }
  };

  // per-lane base pointers into the weight matrix (row n, 16*g k-offset)
  const u16* bp0 = Bw + (size_t)(n0w + (lane & 15)) * K + ((lane >> 4) & 1) * 16;
  const u16* bp1 = bp0 + (size_t)16 * K;

  stage(0, 0);
  __syncthreads();

  for (int kt = 0; kt < nk; ++kt) {
    const int cur = kt & 1;
    if (kt + 1 < nk) {
      stage(kt + 1, cur ^ 1);                       // overlap with WMMAs below
      __builtin_prefetch(bp0 + (kt + 1) * 32, 0, 3); // global_prefetch_b8 (L2 warm)
      __builtin_prefetch(bp1 + (kt + 1) * 32, 0, 3);
    }

    v16bf af[4];
#pragma unroll
    for (int mt = 0; mt < 4; ++mt) af[mt] = load_a_frag(&Als[cur][0], mt * 16, lane);

    if (n0w + 16 <= N) {                            // scalar branch, EXEC untouched
      v16bf bf = load_b_frag(bp0 + kt * 32);
#pragma unroll
      for (int mt = 0; mt < 4; ++mt)
        acc[mt][0] = __builtin_amdgcn_wmma_f32_16x16x32_bf16(
            false, af[mt], false, bf, (short)0, acc[mt][0], false, false);
    }
    if (n0w + 32 <= N) {
      v16bf bf = load_b_frag(bp1 + kt * 32);
#pragma unroll
      for (int mt = 0; mt < 4; ++mt)
        acc[mt][1] = __builtin_amdgcn_wmma_f32_16x16x32_bf16(
            false, af[mt], false, bf, (short)0, acc[mt][1], false, false);
    }
    __syncthreads();
  }

  // epilogue: D vgpr r <-> (M = r + 8*(lane>=16), N = lane&15)
  const int cn = lane & 15, g = (lane >> 4) & 1;
#pragma unroll
  for (int nt = 0; nt < 2; ++nt) {
    if (n0w + nt * 16 + 16 > N) continue;           // scalar
    int n = n0w + nt * 16 + cn;
    float bv = bias[n];
#pragma unroll
    for (int mt = 0; mt < 4; ++mt) {
#pragma unroll
      for (int r = 0; r < 8; ++r) {
        int m = m0 + mt * 16 + r + 8 * g;
        float v = acc[mt][nt][r] + bv;
        if constexpr (EPI == 1) {            // proj layer1: relu -> h (bf16)
          outB0[(size_t)m * N + n] = f2bf(fmaxf(v, 0.0f));
        } else if constexpr (EPI == 2) {     // proj layer2: z0 (f32) / silu (bf16)
          int o = m & 127;
          if (o == 127) outF[(size_t)(m >> 7) * 256 + n] = v;
          else          outB0[(size_t)m * 256 + n] = f2bf(silu_f(v));
        } else {                             // gen: route delta / Bm / Cm
          if (n < 64)        outF [(size_t)m * 64    + n]          = v;
          else if (n < 8256) outB0[(size_t)m * 8192  + (n - 64)]   = f2bf(v);
          else               outB1[(size_t)m * 16384 + (n - 8256)] = f2bf(v);
        }
      }
    }
  }
}

// ---- depthwise conv1d (K=4, pad 3/3, keep last 128) + silu ----------------
__global__ void conv_silu_k(const u16* __restrict__ xall, const float* __restrict__ u,
                            const float* __restrict__ cw, const float* __restrict__ cb,
                            u16* __restrict__ old_) {
  __shared__ float sh[127];
  const int bc = blockIdx.x;
  const int b = bc / 288, c = bc % 288;
  const int j = threadIdx.x;                 // 128 threads = output positions
  if (j < 127) {
    float v;
    if (c < 256) v = bf2f(xall[(size_t)(b * 128 + j) * 256 + c]);
    else         v = silu_f(u[(size_t)b * 8192 + j * 32 + (c - 256)]);
    sh[j] = v;
  }
  __syncthreads();
  float acc = cb[c];
#pragma unroll
  for (int k = 0; k < 4; ++k) {
    int t = j + k - 1;                       // out[j] = conv position j+2
    if (t >= 0 && t < 127) acc += sh[t] * cw[c * 4 + k];
  }
  old_[(size_t)(b * 128 + j) * 288 + c] = f2bf(silu_f(acc));
}

// ---- delta GEMV + softplus + dA/exp + Bm.u0 contraction -------------------
__global__ __launch_bounds__(256) void delta_state_k(
    const float* __restrict__ genD, const float* __restrict__ Wdt,
    const float* __restrict__ bdt, const float* __restrict__ Aarr,
    const u16* __restrict__ Bm, const float* __restrict__ u,
    float* __restrict__ dA, float* __restrict__ dBu)
{
  const int m = blockIdx.x;                  // token (b*128 + l)
  const int b = m >> 7, l = m & 127;
  const int n = threadIdx.x;                 // state index
  __shared__ float sd[64];
  __shared__ float su[32];
  if (n < 64) sd[n] = genD[(size_t)m * 64 + n];
  if (n < 32) su[n] = u[(size_t)b * 8192 + (127 + l) * 32 + n];
  __syncthreads();

  float acc = bdt[n];
#pragma unroll 8
  for (int k = 0; k < 64; ++k) acc += sd[k] * Wdt[n * 64 + k];
  float delta = (acc > 20.0f) ? acc : log1pf(__expf(acc));
  float a  = Aarr[n];
  float an = -((a > 0.0f) ? a : expm1f(a));
  float dAv = __expf(delta * an);
  float scale = (dAv - 1.0f) / an;

  float bs = 0.0f;
  const u16* bp = Bm + (size_t)m * 8192 + n * 32;
#pragma unroll 8
  for (int k = 0; k < 32; ++k) bs += bf2f(bp[k]) * su[k];

  dA [(size_t)m * 256 + n] = dAv;
  dBu[(size_t)m * 256 + n] = scale * bs;
}

// ---- sequential scan: one block per batch chain ---------------------------
__global__ __launch_bounds__(256) void scan_k(
    const float* __restrict__ z0, const float* __restrict__ dA,
    const float* __restrict__ dBu, const u16* __restrict__ Cm,
    const float* __restrict__ x, float* __restrict__ out)
{
  __shared__ float zsh[256];
  __shared__ float psh[4][64];
  __shared__ float lsh[64];
  const int b = blockIdx.x, tid = threadIdx.x;
  const int s = tid & 63, grp = tid >> 6;

  float z = z0[b * 256 + tid];
  float lacc = 0.0f;

  for (int t = 0; t < 128; ++t) {
    size_t m = (size_t)b * 128 + t;
    z = z * dA[m * 256 + tid] + dBu[m * 256 + tid];
    zsh[tid] = z;
    __syncthreads();

    float p = 0.0f;
    const u16* cp = Cm + m * 16384 + (size_t)grp * 64 * 64;
    if (t + 1 < 128) __builtin_prefetch(cp + 16384 + s, 0, 3);  // warm next slab
#pragma unroll 4
    for (int nn = 0; nn < 64; ++nn) p += zsh[grp * 64 + nn] * bf2f(cp[nn * 64 + s]);
    psh[grp][s] = p;
    __syncthreads();

    if (tid < 64) {
      float y = psh[0][tid] + psh[1][tid] + psh[2][tid] + psh[3][tid];
      out[1 + (size_t)(t * 16 + b) * 64 + tid] = y;     // ys[t, b, s]
      float d = y - x[(size_t)b * 16384 + (128 + t) * 64 + tid];
      lacc += d * d;
    }
    __syncthreads();
  }

  if (tid < 64) lsh[tid] = lacc;
  __syncthreads();
  if (tid == 0) {
    float tot = 0.0f;
    for (int i = 0; i < 64; ++i) tot += lsh[i];
    atomicAdd(out, tot * (1.0f / 131072.0f));           // / (B*S) / L
  }
}

// ---------------------------------------------------------------------------
extern "C" void kernel_launch(void* const* d_in, const int* in_sizes, int n_in,
                              void* d_out, int out_size, void* d_ws, size_t ws_size,
                              hipStream_t stream) {
  const float* x   = (const float*)d_in[0];
  const float* u   = (const float*)d_in[1];
  const float* A   = (const float*)d_in[2];
  const float* W0  = (const float*)d_in[3];
  const float* b0  = (const float*)d_in[4];
  const float* W1  = (const float*)d_in[5];
  const float* b1  = (const float*)d_in[6];
  const float* cw  = (const float*)d_in[7];
  const float* cb  = (const float*)d_in[8];
  const float* Wg  = (const float*)d_in[9];
  const float* bg  = (const float*)d_in[10];
  const float* Wdt = (const float*)d_in[11];
  const float* bdt = (const float*)d_in[12];
  float* out = (float*)d_out;

  // workspace bump allocator (~125 MB total)
  char* ws = (char*)d_ws;
  size_t off = 0;
  auto alloc = [&](size_t bytes) -> char* {
    char* p = ws + off;
    off = (off + bytes + 255) & ~(size_t)255;
    return p;
  };
  u16*   W0b  = (u16*)  alloc(512  * 64   * 2);
  u16*   W1b  = (u16*)  alloc(256  * 512  * 2);
  u16*   Wgb  = (u16*)  alloc((size_t)24640 * 288 * 2);
  u16*   xtok = (u16*)  alloc(2048 * 64   * 2);
  u16*   h    = (u16*)  alloc(2048 * 512  * 2);
  u16*   xall = (u16*)  alloc(2048 * 256  * 2);
  float* z0   = (float*)alloc(16   * 256  * 4);
  u16*   oldb = (u16*)  alloc(2048 * 288  * 2);
  float* genD = (float*)alloc(2048 * 64   * 4);
  u16*   Bm   = (u16*)  alloc((size_t)2048 * 8192  * 2);
  u16*   Cm   = (u16*)  alloc((size_t)2048 * 16384 * 2);
  float* dA   = (float*)alloc(2048 * 256  * 4);
  float* dBu  = (float*)alloc(2048 * 256  * 4);

  // 0) precision conversion (weights once per call; all L2-resident after)
  f2bf_k<<<(512 * 64 + 255) / 256, 256, 0, stream>>>(W0, W0b, 512 * 64);
  f2bf_k<<<(256 * 512 + 255) / 256, 256, 0, stream>>>(W1, W1b, 256 * 512);
  f2bf_k<<<(24640 * 288 + 255) / 256, 256, 0, stream>>>(Wg, Wgb, 24640 * 288);
  xtok_k<<<(2048 * 64 + 255) / 256, 256, 0, stream>>>(x, xtok);

  // 1) proj layer1: h = relu(xtok @ W0^T + b0)       (2048 x 512, K=64)
  gemm_bf16_k<1><<<dim3(2, 32), 256, 0, stream>>>(
      xtok, W0b, b0, nullptr, h, nullptr, 2048, 512, 64);
  // 2) proj layer2: z0 / silu(xall)                  (2048 x 256, K=512)
  gemm_bf16_k<2><<<dim3(1, 32), 256, 0, stream>>>(
      h, W1b, b1, z0, xall, nullptr, 2048, 256, 512);
  // 3) depthwise conv + silu -> old                  (B*288 channels)
  conv_silu_k<<<16 * 288, 128, 0, stream>>>(xall, u, cw, cb, oldb);
  // 4) THE GEMM: gen = old @ Wg^T + bg, routed to delta/Bm/Cm (2048 x 24640, K=288)
  gemm_bf16_k<3><<<dim3(97, 32), 256, 0, stream>>>(
      oldb, Wgb, bg, genD, Bm, Cm, 2048, 24640, 288);
  // 5) delta -> softplus -> dA, and dBu = scale * (Bm . u0)
  delta_state_k<<<2048, 256, 0, stream>>>(genD, Wdt, bdt, A, Bm, u, dA, dBu);
  // 6) scan
  hipMemsetAsync(d_out, 0, sizeof(float), stream);
  scan_k<<<16, 256, 0, stream>>>(z0, dA, dBu, Cm, x, out);
}
